// Sinkhorn_custom_17978733101130
// MI455X (gfx1250) — matrix operand order repaired and checked by reading, not compiled, not run on hardware
//
#include <hip/hip_runtime.h>
#include <hip/hip_bf16.h>
#include <stdint.h>

typedef __attribute__((ext_vector_type(16))) _Float16 v16h;
typedef __attribute__((ext_vector_type(8)))  _Float16 v8h;
typedef __attribute__((ext_vector_type(8)))  float    v8f;
typedef __attribute__((ext_vector_type(4)))  float    v4f;

#define BS 32
#define NN 1024
#define MM 1024
#define EPS 0.1f
#define MAX_ITER 200

// ---------------------------------------------------------------------------
// Kernel A: fused prep.  Per 64x64 tile: G16 = (f16)exp(-C/eps) written
// row-major, and GT16 (transposed copy) written via an LDS-staged tiled
// transpose so BOTH iteration phases get contiguous b128 loads.
// HBM traffic: 128 MB read + 2x64 MB write  ->  ~11 us at 23.3 TB/s.
// ---------------------------------------------------------------------------
__global__ void __launch_bounds__(256)
sk_prep(const float* __restrict__ C,
        _Float16* __restrict__ G, _Float16* __restrict__ GT) {
  __shared__ _Float16 tile[64][72];            // +8 halves pad: no bank conflicts
  const int t  = threadIdx.x;
  const int tj = t & 7;                        // 8-half chunk within the tile row
  const int ti = t >> 3;                       // 0..31
  int blk = blockIdx.x;
  const int jt = blk & 15; blk >>= 4;
  const int it = blk & 15; blk >>= 4;
  const int b  = blk;
  const int i0 = it * 64, j0 = jt * 64;
  const float neg_inv_eps = -1.0f / EPS;
  const size_t base = ((size_t)b << 20);       // b * 1024 * 1024

  #pragma unroll
  for (int rr = 0; rr < 2; ++rr) {
    const int r = ti + rr * 32;                // tile row (i direction)
    const float* cp = C + base + (size_t)(i0 + r) * MM + j0 + tj * 8;
    v4f c0 = *(const v4f*)cp;
    v4f c1 = *(const v4f*)(cp + 4);
    v8h g;
    g[0] = (_Float16)__expf(c0[0] * neg_inv_eps);
    g[1] = (_Float16)__expf(c0[1] * neg_inv_eps);
    g[2] = (_Float16)__expf(c0[2] * neg_inv_eps);
    g[3] = (_Float16)__expf(c0[3] * neg_inv_eps);
    g[4] = (_Float16)__expf(c1[0] * neg_inv_eps);
    g[5] = (_Float16)__expf(c1[1] * neg_inv_eps);
    g[6] = (_Float16)__expf(c1[2] * neg_inv_eps);
    g[7] = (_Float16)__expf(c1[3] * neg_inv_eps);
    *(v8h*)(G + base + (size_t)(i0 + r) * MM + j0 + tj * 8) = g;
    #pragma unroll
    for (int s = 0; s < 8; ++s) tile[r][tj * 8 + s] = g[s];
  }
  __syncthreads();
  #pragma unroll
  for (int rr = 0; rr < 2; ++rr) {
    const int jr = ti + rr * 32;               // row of GT tile (j direction)
    v8h g;
    #pragma unroll
    for (int s = 0; s < 8; ++s) g[s] = tile[tj * 8 + s][jr];
    *(v8h*)(GT + base + (size_t)(j0 + jr) * NN + i0 + tj * 8) = g;
  }
}

// ---------------------------------------------------------------------------
// Kernel B: persistent Sinkhorn iterations.  One workgroup (32 waves) per
// batch; G16+GT16 (4 MB per batch, 128 MB total) stay L2-resident across all
// 400 passes.  u,v live in LDS; workgroup barrier between phases.
//
// Phase 1 (u = mu / (G v)):    D = A*B, A = 16x32 row-tile of G,
//   B[k][n] = v[k] broadcast  -> every column of D is the rowsum.
// Phase 2 (v = nu / (u^T G)):  D = A*B, A[m][k] = u[k] broadcast,
//   B[k][n] = GT[j0+n][k0+k]  -> lane n reads 16 contiguous halves of GT;
//   every row of D is the colsum.
// ---------------------------------------------------------------------------
__global__ void __launch_bounds__(1024, 1)
sk_iter(const _Float16* __restrict__ G, const _Float16* __restrict__ GT,
        float* __restrict__ Uout, float* __restrict__ Vout) {
  __shared__ __align__(32) _Float16 vh[MM];
  __shared__ __align__(32) _Float16 uh[NN];
  __shared__ float uf[NN];
  __shared__ float vf[MM];

  const int tid  = threadIdx.x;
  const int lane = tid & 31;
  const int wave = tid >> 5;
  const int hi   = lane >> 4;   // 0: lanes 0-15, 1: lanes 16-31
  const int lo   = lane & 15;
  const int b    = blockIdx.x;
  const _Float16* Gb  = G  + ((size_t)b << 20);
  const _Float16* GTb = GT + ((size_t)b << 20);

  vh[tid] = (_Float16)(1.0f / MM);             // v0 = 1/m
  __syncthreads();

  for (int it = 0; it < MAX_ITER; ++it) {
    // ------- phase 1: rowsums; wave owns rows [32w, 32w+32) ---------------
    #pragma unroll
    for (int t = 0; t < 2; ++t) {
      const int i0 = wave * 32 + t * 16;
      const _Float16* rowp = Gb + (size_t)(i0 + lo) * MM;
      v8f acc = {};
      #pragma unroll 4
      for (int k0 = 0; k0 < MM; k0 += 32) {
        __builtin_prefetch(rowp + k0 + 128, 0, 3);   // WGP-scope prefetch
        // A-fragment: contiguous K layout (K 0..15 / 16..31), matches B
        v8h a0 = *(const v8h*)(rowp + k0 + hi * 16);
        v8h a1 = *(const v8h*)(rowp + k0 + hi * 16 + 8);
        v16h A = __builtin_shufflevector(a0, a1,
                   0,1,2,3,4,5,6,7,8,9,10,11,12,13,14,15);
        // B-fragment: B[k][n] = v[k0+k]
        v16h B = *(const v16h*)(vh + k0 + hi * 16);
        acc = __builtin_amdgcn_wmma_f32_16x16x32_f16(
                  false, A, false, B, (short)0, acc, false, false);
      }
      if (lo == 0) {                           // lane n=0 (+16) holds rows 0-7 (8-15)
        #pragma unroll
        for (int r = 0; r < 8; ++r) {
          const int   row  = i0 + hi * 8 + r;
          const float uval = (1.0f / NN) * __builtin_amdgcn_rcpf(acc[r]);
          uh[row] = (_Float16)uval;
          uf[row] = uval;
        }
      }
    }
    __syncthreads();

    // ------- phase 2: colsums; wave owns cols [32w, 32w+32) ---------------
    #pragma unroll
    for (int t = 0; t < 2; ++t) {
      const int j0 = wave * 32 + t * 16;
      const _Float16* colp = GTb + (size_t)(j0 + lo) * NN;
      v8f acc = {};
      #pragma unroll 4
      for (int k0 = 0; k0 < NN; k0 += 32) {
        __builtin_prefetch(colp + k0 + 128, 0, 3);
        // A-fragment: A[m][k] = u[k0+k] (identical rows)
        v8h a0 = *(const v8h*)(uh + k0 + hi * 16);
        v8h a1 = *(const v8h*)(uh + k0 + hi * 16 + 8);
        v16h A = __builtin_shufflevector(a0, a1,
                   0,1,2,3,4,5,6,7,8,9,10,11,12,13,14,15);
        // B-fragment: lane n=lo holds GT[j0+n][k0 + hi*16 .. +15], contiguous
        v16h B = *(const v16h*)(colp + k0 + hi * 16);
        acc = __builtin_amdgcn_wmma_f32_16x16x32_f16(
                  false, A, false, B, (short)0, acc, false, false);
      }
      if (hi == 0) {                           // every row of D identical
        const float vval = (1.0f / MM) * __builtin_amdgcn_rcpf(acc[0]);
        vh[j0 + lo] = (_Float16)vval;
        vf[j0 + lo] = vval;
      }
    }
    __syncthreads();
  }

  Uout[b * NN + tid] = uf[tid];
  Vout[b * MM + tid] = vf[tid];
}

// ---------------------------------------------------------------------------
// Kernel C: Gamma*n = u[i] * exp(-C/eps) * v[j] * n, fp32 throughout.
// 256 MB of HBM traffic -> ~11 us.
// ---------------------------------------------------------------------------
__global__ void sk_final(const float* __restrict__ C,
                         const float* __restrict__ U,
                         const float* __restrict__ V,
                         float* __restrict__ out) {
  const float neg_inv_eps = -1.0f / EPS;
  size_t idx    = (size_t)blockIdx.x * blockDim.x + threadIdx.x;
  size_t stride = (size_t)gridDim.x * blockDim.x;
  const size_t total4 = (size_t)BS * NN * MM / 4;
  for (size_t e = idx; e < total4; e += stride) {
    const size_t g = e * 4;
    const int j  = (int)(g & (MM - 1));
    const int i  = (int)((g >> 10) & (NN - 1));
    const int bb = (int)(g >> 20);
    v4f c  = ((const v4f*)C)[e];
    v4f vv = *(const v4f*)(V + bb * MM + j);
    const float us = U[bb * NN + i] * (float)NN;
    v4f o;
    o[0] = us * __expf(c[0] * neg_inv_eps) * vv[0];
    o[1] = us * __expf(c[1] * neg_inv_eps) * vv[1];
    o[2] = us * __expf(c[2] * neg_inv_eps) * vv[2];
    o[3] = us * __expf(c[3] * neg_inv_eps) * vv[3];
    ((v4f*)out)[e] = o;
  }
}

// ---------------------------------------------------------------------------
extern "C" void kernel_launch(void* const* d_in, const int* in_sizes, int n_in,
                              void* d_out, int out_size, void* d_ws, size_t ws_size,
                              hipStream_t stream) {
  (void)in_sizes; (void)n_in; (void)out_size; (void)ws_size;
  const float* C = (const float*)d_in[0];
  float* out = (float*)d_out;

  // workspace: [ G16 : 64 MB ][ GT16 : 64 MB ][ U : 128 KB ][ V : 128 KB ]
  const size_t gbytes = (size_t)BS * NN * MM * sizeof(_Float16);
  _Float16* G  = (_Float16*)d_ws;
  _Float16* GT = (_Float16*)((char*)d_ws + gbytes);
  float* U = (float*)((char*)d_ws + 2 * gbytes);
  float* V = U + BS * NN;

  sk_prep <<<BS * 16 * 16, 256, 0, stream>>>(C, G, GT);
  sk_iter <<<BS, 1024, 0, stream>>>(G, GT, U, V);
  sk_final<<<8192, 256, 0, stream>>>(C, U, V, out);
}
